// MolecularGCN_60722247631257
// MI455X (gfx1250) — matrix operand order repaired and checked by reading, not compiled, not run on hardware
//
#include <hip/hip_runtime.h>
#include <hip/hip_bf16.h>

// ---------------- problem constants (match reference) ----------------
#define NN   100000      // nodes
#define EE   1600000     // edges
#define F_IN 128
#define HH   256
#define OUTF 128
#define GG   2048
#define BN_EPS 1e-5f

// ---------------- vector types for WMMA ----------------
typedef _Float16 v16h __attribute__((ext_vector_type(16)));
typedef _Float16 v8h  __attribute__((ext_vector_type(8)));
typedef float    v8f  __attribute__((ext_vector_type(8)));

// =====================================================================
// Degree / normalization
// =====================================================================
__global__ void k_deg_init(float* __restrict__ deg) {
    int i = blockIdx.x * blockDim.x + threadIdx.x;
    if (i < NN) deg[i] = 1.0f;               // +1 self loop
}
__global__ void k_deg_acc(const int* __restrict__ col, float* __restrict__ deg) {
    int e = blockIdx.x * blockDim.x + threadIdx.x;
    if (e < EE) atomicAdd(&deg[col[e]], 1.0f);
}
__global__ void k_dinv(float* __restrict__ deg) {
    int i = blockIdx.x * blockDim.x + threadIdx.x;
    if (i < NN) deg[i] = rsqrtf(deg[i]);     // in place: deg -> dinv
}

// =====================================================================
// f32 -> f16 convert (activations), and weight transpose+convert
// =====================================================================
__global__ void k_f32_to_f16(const float* __restrict__ x, _Float16* __restrict__ y,
                             long long total) {
    long long i = blockIdx.x * (long long)blockDim.x + threadIdx.x;
    if (i < total) y[i] = (_Float16)x[i];
}
// W [K, Nf] (row-major) -> Wt [Nf, K] f16
__global__ void k_w_t_f16(const float* __restrict__ W, _Float16* __restrict__ Wt,
                          int K, int Nf) {
    int i = blockIdx.x * blockDim.x + threadIdx.x;
    if (i >= K * Nf) return;
    int k = i / Nf, n = i % Nf;
    Wt[(size_t)n * K + k] = (_Float16)W[i];
}

// =====================================================================
// WMMA GEMM: C[M,Nf] = A[M,K](f16) * Bt[Nf,K](f16)^T, f32 accumulate.
// One wave (32 lanes) computes one 16x16 tile. M%16==0, K%32==0, Nf%16==0.
// Fragment layouts follow CDNA5 ISA 7.12.2 (16-bit A 16x32, B 32x16).
// =====================================================================
__global__ __launch_bounds__(32) void k_gemm_f16_wmma(
    const _Float16* __restrict__ A, const _Float16* __restrict__ Bt,
    float* __restrict__ C, int K, int Nf)
{
    const int lane = threadIdx.x & 31;
    const bool hi  = lane >= 16;
    const int l15  = lane & 15;

    const size_t row  = (size_t)blockIdx.x * 16 + l15;   // A row for this lane
    const int    colN = blockIdx.y * 16 + l15;           // B/C column for this lane

    const _Float16* arow = A  + row * (size_t)K;
    const _Float16* bcol = Bt + (size_t)colN * K + (hi ? 8 : 0);

    v8f acc = {};
    for (int k0 = 0; k0 < K; k0 += 32) {
        if (k0 + 32 < K) __builtin_prefetch(arow + k0 + 32, 0, 0);
        // A 16x32 f16: lanes<16 hold K = [k0,k0+8) and [k0+16,k0+24),
        //              lanes>=16 hold K = [k0+8,k0+16) and [k0+24,k0+32)
        const v8h* ap = (const v8h*)(arow + k0);
        v8h a0 = ap[hi ? 1 : 0];
        v8h a1 = ap[hi ? 3 : 2];
        // B 32x16 f16: lane holds column colN; lanes<16: K=[k0,k0+8) and [k0+16,k0+24), lanes>=16: K=[k0+8,k0+16) and [k0+24,k0+32)
        const v8h* bp = (const v8h*)(bcol + k0);
        v8h b0 = bp[0];
        v8h b1 = bp[2];
        v16h af = __builtin_shufflevector(a0, a1, 0,1,2,3,4,5,6,7,8,9,10,11,12,13,14,15);
        v16h bf = __builtin_shufflevector(b0, b1, 0,1,2,3,4,5,6,7,8,9,10,11,12,13,14,15);
        acc = __builtin_amdgcn_wmma_f32_16x16x32_f16(
            /*neg_a=*/false, af, /*neg_b=*/false, bf,
            /*c_mod=*/(short)0, acc, /*reuse_a=*/false, /*reuse_b=*/false);
    }
    // C/D layout: lane holds column colN; VGPR r holds row r (+8 for hi lanes)
    const int rbase = blockIdx.x * 16 + (hi ? 8 : 0);
    #pragma unroll
    for (int r = 0; r < 8; ++r)
        C[(size_t)(rbase + r) * Nf + colN] = acc[r];
}

// =====================================================================
// GCN aggregation
// =====================================================================
// self-loop messages: agg = h * dinv^2   (per node)
__global__ void k_self_init(const float* __restrict__ h, const float* __restrict__ dinv,
                            float* __restrict__ agg, int F, long long total) {
    long long i = blockIdx.x * (long long)blockDim.x + threadIdx.x;
    if (i >= total) return;
    int n = (int)(i / F);
    float d = dinv[n];
    agg[i] = h[i] * d * d;
}
// edge messages: agg[col] += h[row] * dinv[row]*dinv[col]; float4 per thread
__global__ void k_edge_agg(const int* __restrict__ row, const int* __restrict__ col,
                           const float* __restrict__ dinv, const float* __restrict__ h,
                           float* __restrict__ agg, int F, long long total) {
    long long i = blockIdx.x * (long long)blockDim.x + threadIdx.x;
    if (i >= total) return;
    int qper = F >> 2;
    int e = (int)(i / qper);
    int q = ((int)(i % qper)) << 2;
    int r = row[e], c = col[e];
    float nrm = dinv[r] * dinv[c];
    const float4 v = *(const float4*)(h + (size_t)r * F + q);
    float* dst = agg + (size_t)c * F + q;
    atomicAdd(dst + 0, v.x * nrm);
    atomicAdd(dst + 1, v.y * nrm);
    atomicAdd(dst + 2, v.z * nrm);
    atomicAdd(dst + 3, v.w * nrm);
}

// =====================================================================
// BatchNorm (training stats over N nodes)
// =====================================================================
__global__ __launch_bounds__(256) void k_bn_stats(const float* __restrict__ X, int F,
                                                  float* __restrict__ mean,
                                                  float* __restrict__ var) {
    __shared__ float s1[256], s2[256];
    int f = blockIdx.x;
    float sum = 0.f, sq = 0.f;
    for (int i = threadIdx.x; i < NN; i += 256) {
        float v = X[(size_t)i * F + f];
        sum += v; sq += v * v;
    }
    s1[threadIdx.x] = sum; s2[threadIdx.x] = sq;
    __syncthreads();
    for (int s = 128; s > 0; s >>= 1) {
        if (threadIdx.x < s) { s1[threadIdx.x] += s1[threadIdx.x + s];
                               s2[threadIdx.x] += s2[threadIdx.x + s]; }
        __syncthreads();
    }
    if (threadIdx.x == 0) {
        float m = s1[0] / (float)NN;
        mean[f] = m;
        var[f]  = s2[0] / (float)NN - m * m;
    }
}
// BN + ReLU, emit f16 for next layer's WMMA input
__global__ void k_bn_relu_f16(const float* __restrict__ X, const float* __restrict__ mean,
                              const float* __restrict__ var, const float* __restrict__ g,
                              const float* __restrict__ b, _Float16* __restrict__ Y,
                              int F, long long total) {
    long long i = blockIdx.x * (long long)blockDim.x + threadIdx.x;
    if (i >= total) return;
    int f = (int)(i % F);
    float y = g[f] * (X[i] - mean[f]) * rsqrtf(var[f] + BN_EPS) + b[f];
    Y[i] = (_Float16)fmaxf(y, 0.f);
}
// BN only, f32 out (final layer, no ReLU)
__global__ void k_bn_f32(const float* __restrict__ X, const float* __restrict__ mean,
                         const float* __restrict__ var, const float* __restrict__ g,
                         const float* __restrict__ b, float* __restrict__ Y,
                         int F, long long total) {
    long long i = blockIdx.x * (long long)blockDim.x + threadIdx.x;
    if (i >= total) return;
    int f = (int)(i % F);
    Y[i] = g[f] * (X[i] - mean[f]) * rsqrtf(var[f] + BN_EPS) + b[f];
}

// =====================================================================
// Global mean pool over graphs
// =====================================================================
__global__ void k_zero_f32(float* __restrict__ p, long long total) {
    long long i = blockIdx.x * (long long)blockDim.x + threadIdx.x;
    if (i < total) p[i] = 0.f;
}
__global__ void k_pool_acc(const float* __restrict__ h, const int* __restrict__ batch,
                           float* __restrict__ out, long long total) {
    long long i = blockIdx.x * (long long)blockDim.x + threadIdx.x;
    if (i >= total) return;
    int n = (int)(i / OUTF), f = (int)(i % OUTF);
    atomicAdd(&out[(size_t)batch[n] * OUTF + f], h[i]);
}
__global__ void k_cnt_acc(const int* __restrict__ batch, float* __restrict__ cnt) {
    int n = blockIdx.x * blockDim.x + threadIdx.x;
    if (n < NN) atomicAdd(&cnt[batch[n]], 1.0f);
}
__global__ void k_pool_div(float* __restrict__ out, const float* __restrict__ cnt,
                           long long total) {
    long long i = blockIdx.x * (long long)blockDim.x + threadIdx.x;
    if (i >= total) return;
    out[i] /= fmaxf(cnt[i / OUTF], 1.0f);
}

// =====================================================================
// Launcher
// =====================================================================
static inline int blks(long long total, int bs) { return (int)((total + bs - 1) / bs); }

extern "C" void kernel_launch(void* const* d_in, const int* in_sizes, int n_in,
                              void* d_out, int out_size, void* d_ws, size_t ws_size,
                              hipStream_t stream) {
    (void)in_sizes; (void)n_in; (void)out_size; (void)ws_size;
    const float* x     = (const float*)d_in[0];
    const int*   row   = (const int*)  d_in[1];
    const int*   col   = (const int*)  d_in[2];
    const int*   batch = (const int*)  d_in[3];
    const float* W1 = (const float*)d_in[4];
    const float* g1 = (const float*)d_in[6];  const float* be1 = (const float*)d_in[7];
    const float* W2 = (const float*)d_in[8];
    const float* g2 = (const float*)d_in[10]; const float* be2 = (const float*)d_in[11];
    const float* W3 = (const float*)d_in[12];
    const float* g3 = (const float*)d_in[14]; const float* be3 = (const float*)d_in[15];
    float* out = (float*)d_out;
    // NOTE: per-feature biases b1/b2/b3 are mathematically cancelled by the
    // immediately-following BatchNorm (shift invariance), so they are skipped.

    // ---- workspace partition ----
    char* p = (char*)d_ws;
    size_t off = 0;
    auto take = [&](size_t bytes) { void* q = p + off; off += (bytes + 255) & ~(size_t)255; return q; };
    _Float16* X16  = (_Float16*)take((size_t)NN * HH * 2);   // f16 activations (max N x 256)
    _Float16* WT   = (_Float16*)take((size_t)HH * HH * 2);   // transposed f16 weights
    float*    bufA = (float*)   take((size_t)NN * HH * 4);   // GEMM output h
    float*    bufB = (float*)   take((size_t)NN * HH * 4);   // aggregation output
    float*    dinv = (float*)   take((size_t)NN * 4);
    float*    mean = (float*)   take((size_t)HH * 4);
    float*    var  = (float*)   take((size_t)HH * 4);
    float*    cnt  = (float*)   take((size_t)GG * 4);

    const int BS = 256;

    // ---- degrees / symmetric normalization (shared by all layers) ----
    k_deg_init<<<blks(NN, BS), BS, 0, stream>>>(dinv);
    k_deg_acc <<<blks(EE, BS), BS, 0, stream>>>(col, dinv);
    k_dinv    <<<blks(NN, BS), BS, 0, stream>>>(dinv);

    // ================= Layer 1: [N,128] -> [N,256], BN + ReLU =================
    {
        const int K = F_IN, Nf = HH;
        long long nk = (long long)NN * K, nf = (long long)NN * Nf;
        k_f32_to_f16<<<blks(nk, BS), BS, 0, stream>>>(x, X16, nk);
        k_w_t_f16<<<blks((long long)K * Nf, BS), BS, 0, stream>>>(W1, WT, K, Nf);
        k_gemm_f16_wmma<<<dim3(NN / 16, Nf / 16), 32, 0, stream>>>(X16, WT, bufA, K, Nf);
        k_self_init<<<blks(nf, BS), BS, 0, stream>>>(bufA, dinv, bufB, Nf, nf);
        long long tq = (long long)EE * (Nf >> 2);
        k_edge_agg<<<blks(tq, BS), BS, 0, stream>>>(row, col, dinv, bufA, bufB, Nf, tq);
        k_bn_stats<<<Nf, 256, 0, stream>>>(bufB, Nf, mean, var);
        k_bn_relu_f16<<<blks(nf, BS), BS, 0, stream>>>(bufB, mean, var, g1, be1, X16, Nf, nf);
    }
    // ================= Layer 2: [N,256] -> [N,256], BN + ReLU =================
    {
        const int K = HH, Nf = HH;
        long long nf = (long long)NN * Nf;
        k_w_t_f16<<<blks((long long)K * Nf, BS), BS, 0, stream>>>(W2, WT, K, Nf);
        k_gemm_f16_wmma<<<dim3(NN / 16, Nf / 16), 32, 0, stream>>>(X16, WT, bufA, K, Nf);
        k_self_init<<<blks(nf, BS), BS, 0, stream>>>(bufA, dinv, bufB, Nf, nf);
        long long tq = (long long)EE * (Nf >> 2);
        k_edge_agg<<<blks(tq, BS), BS, 0, stream>>>(row, col, dinv, bufA, bufB, Nf, tq);
        k_bn_stats<<<Nf, 256, 0, stream>>>(bufB, Nf, mean, var);
        k_bn_relu_f16<<<blks(nf, BS), BS, 0, stream>>>(bufB, mean, var, g2, be2, X16, Nf, nf);
    }
    // ================= Layer 3: [N,256] -> [N,128], BN (no ReLU) =================
    {
        const int K = HH, Nf = OUTF;
        long long nf = (long long)NN * Nf;
        k_w_t_f16<<<blks((long long)K * Nf, BS), BS, 0, stream>>>(W3, WT, K, Nf);
        k_gemm_f16_wmma<<<dim3(NN / 16, Nf / 16), 32, 0, stream>>>(X16, WT, bufA, K, Nf);
        k_self_init<<<blks(nf, BS), BS, 0, stream>>>(bufA, dinv, bufB, Nf, nf);
        long long tq = (long long)EE * (Nf >> 2);
        k_edge_agg<<<blks(tq, BS), BS, 0, stream>>>(row, col, dinv, bufA, bufB, Nf, tq);
        k_bn_stats<<<Nf, 256, 0, stream>>>(bufB, Nf, mean, var);
        k_bn_f32<<<blks(nf, BS), BS, 0, stream>>>(bufB, mean, var, g3, be3, bufA, Nf, nf);
    }
    // ================= global mean pool =================
    {
        long long go = (long long)GG * OUTF;
        long long no = (long long)NN * OUTF;
        k_zero_f32<<<blks(go, BS), BS, 0, stream>>>(out, go);
        k_zero_f32<<<blks(GG, BS), BS, 0, stream>>>(cnt, GG);
        k_pool_acc<<<blks(no, BS), BS, 0, stream>>>(bufA, batch, out, no);
        k_cnt_acc <<<blks(NN, BS), BS, 0, stream>>>(batch, cnt);
        k_pool_div<<<blks(go, BS), BS, 0, stream>>>(out, cnt, go);
    }
}
